// CLSTMCell_51677046505499
// MI455X (gfx1250) — matrix operand order repaired and checked by reading, not compiled, not run on hardware
//
#include <hip/hip_runtime.h>
#include <hip/hip_bf16.h>
#include <math.h>

// ---------------------------------------------------------------------------
// Complex LSTM cell, MI455X (gfx1250, wave32, WMMA).
// Z = X(4096x4096) @ W(4096x8192) as bf16 WMMA (f32 acc), fused gate epilogue.
// ws layout: [X_bf16 : 32MB][Wt_bf16 : 64MB]; both stay resident in 192MB L2.
// ---------------------------------------------------------------------------

typedef __attribute__((ext_vector_type(16))) __bf16 v16bf;
typedef __attribute__((ext_vector_type(8)))  __bf16 v8bf;
typedef __attribute__((ext_vector_type(8)))  float  v8f;

#define BATCH 4096
#define DK    4096            // shared K dimension (2D = 2U = 4096)
#define BM    128
#define KSTEP 64              // K per LDS stage (two 32-K WMMA sub-steps)
#define NS    (DK / KSTEP)    // 64 stages
#define AROW  72              // 64 elems + 8 pad -> conflict-free frag reads

// ---- pack X = [inputs | h_tm1] -> bf16 ------------------------------------
__global__ __launch_bounds__(256) void pack_x(const float* __restrict__ inputs,
                                              const float* __restrict__ h_tm1,
                                              __bf16* __restrict__ X) {
  size_t idx = (size_t)blockIdx.x * 256 + threadIdx.x;   // over 4096*1024
  int b  = (int)(idx >> 10);
  int k  = (int)(idx & 1023) * 4;
  const float* src = (k < 2048) ? (inputs + (size_t)b * 2048 + k)
                                : (h_tm1  + (size_t)b * 2048 + (k - 2048));
  float4 v = *(const float4*)src;
  __bf16* dst = X + (size_t)b * DK + k;
  dst[0] = (__bf16)v.x; dst[1] = (__bf16)v.y;
  dst[2] = (__bf16)v.z; dst[3] = (__bf16)v.w;
}

// ---- pack Wt[n,k] (8192 x 4096, bf16), LDS-tiled transpose with signs -----
__global__ __launch_bounds__(256) void pack_wt(const float* __restrict__ rk,
                                               const float* __restrict__ ik,
                                               const float* __restrict__ rrk,
                                               const float* __restrict__ irk,
                                               __bf16* __restrict__ Wt) {
  __shared__ float tile[32][33];
  const int k0  = blockIdx.x * 32;       // 0..4095
  const int n0  = blockIdx.y * 32;       // 0..8191
  const int ri  = n0 >> 12;              // 0 = zr cols, 1 = zi cols
  const int seg = k0 >> 10;              // which source matrix along K
  const int kr0 = k0 & 1023;
  const int c0  = n0 & 4095;
  const float* src; float sgn = 1.0f;
  if (ri == 0) {                         // zr = xr@Wr + xi@Wi + hr@Rr + hi@Ri
    src = (seg == 0) ? rk : (seg == 1) ? ik : (seg == 2) ? rrk : irk;
  } else {                               // zi = -xr@Wi + xi@Wr - hr@Ri + hi@Rr
    if      (seg == 0) { src = ik;  sgn = -1.0f; }
    else if (seg == 1) { src = rk; }
    else if (seg == 2) { src = irk; sgn = -1.0f; }
    else               { src = rrk; }
  }
  const int tx = threadIdx.x & 31, ty = threadIdx.x >> 5;
  for (int yy = 0; yy < 4; ++yy) {
    int kk = ty * 4 + yy;
    tile[kk][tx] = sgn * src[(size_t)(kr0 + kk) * 4096 + c0 + tx]; // coalesced
  }
  __syncthreads();
  for (int yy = 0; yy < 4; ++yy) {
    int nn = ty * 4 + yy;
    Wt[(size_t)(n0 + nn) * DK + k0 + tx] = (__bf16)tile[tx][nn];   // coalesced
  }
}

// ---- fused GEMM + gates ---------------------------------------------------
union SharedU {
  __bf16 a[2][BM * AROW];                // 2 x 18.4 KB double-buffered A tile
  float  z[8 * 4 * 256];                 // 32 KB gate staging (after K loop)
};

union BF16x16 { v16bf v; v8bf h[2]; };

__global__ __launch_bounds__(256) void clstm_gemm(const __bf16* __restrict__ X,
                                                  const __bf16* __restrict__ Wt,
                                                  const float* __restrict__ c_tm1,
                                                  const float* __restrict__ rbias,
                                                  const float* __restrict__ ibias,
                                                  float* __restrict__ out) {
  __shared__ SharedU sm;

  const int t    = threadIdx.x;
  const int lane = t & 31;
  const int w    = t >> 5;               // wave 0..7
  const int ri   = w >> 2;               // 0: zr, 1: zi
  const int gate = w & 3;                // i, f, ctilde, o
  const int m0   = blockIdx.x * BM;      // batch-row tile
  const int j0   = blockIdx.y * 16;      // output-channel slice (within U)
  const int nbase = ri * 4096 + gate * 1024 + j0;

  v8f acc[8] = {};

  // B fragments: Wt row (= output column) per lane 0..15, K-contiguous.
  // contiguous k order (same as A): lanes 0-15 hold K[0:16), lanes 16-31 K[16:32).
  const __bf16* gB = Wt + (size_t)(nbase + (lane & 15)) * DK + ((lane >> 4) * 16);

  // cooperative A staging: thread -> (row, 32-elem half) of the 128x64 tile
  const int arow  = t >> 1;
  const int acol  = (t & 1) * 32;
  const __bf16* gA = X + (size_t)(m0 + arow) * DK + acol;
  const int sAoff = arow * AROW + acol;

  // A fragment addressing (contiguous k order, matching B):
  // lanes 0-15: K[0:16); lanes 16-31: K[16:32)
  const int fr  = lane & 15;
  const int fc0 = (lane >> 4) * 16;

  // ---- prologue: stage tile 0, fetch B frags for step 0 -------------------
  v8bf areg[4];
  BF16x16 bcur[2], bnext[2];
#pragma unroll
  for (int q = 0; q < 4; ++q) areg[q] = *(const v8bf*)(gA + q * 8);
  bcur[0].h[0] = *(const v8bf*)(gB);      bcur[0].h[1] = *(const v8bf*)(gB + 8);
  bcur[1].h[0] = *(const v8bf*)(gB + 32); bcur[1].h[1] = *(const v8bf*)(gB + 40);
  {
    __bf16* sA0 = sm.a[0] + sAoff;
#pragma unroll
    for (int q = 0; q < 4; ++q) *(v8bf*)(sA0 + q * 8) = areg[q];
  }

#pragma unroll 2
  for (int s = 0; s < NS; ++s) {
    const __bf16* aTile = sm.a[s & 1];
    const bool haveNext = (s + 1) < NS;

    // global prefetch of next tile / next B frags (overlaps this step's math)
    if (haveNext) {
      const __bf16* gAn = gA + (s + 1) * KSTEP;
#pragma unroll
      for (int q = 0; q < 4; ++q) areg[q] = *(const v8bf*)(gAn + q * 8);
      const __bf16* gBn = gB + (s + 1) * KSTEP;
      bnext[0].h[0] = *(const v8bf*)(gBn);      bnext[0].h[1] = *(const v8bf*)(gBn + 8);
      bnext[1].h[0] = *(const v8bf*)(gBn + 32); bnext[1].h[1] = *(const v8bf*)(gBn + 40);
    }

    __syncthreads();                     // lds[s&1] stores visible everywhere

#pragma unroll
    for (int hk = 0; hk < 2; ++hk) {
      BF16x16 ua[8];
      const __bf16* base = aTile + hk * 32 + fc0;
#pragma unroll
      for (int m = 0; m < 8; ++m) {      // batch all ds_loads first
        const __bf16* ap = base + (m * 16 + fr) * AROW;
        ua[m].h[0] = *(const v8bf*)(ap);
        ua[m].h[1] = *(const v8bf*)(ap + 8);
      }
#pragma unroll
      for (int m = 0; m < 8; ++m)        // then the WMMA chain
        acc[m] = __builtin_amdgcn_wmma_f32_16x16x32_bf16(
            false, ua[m].v, false, bcur[hk].v, (short)0, acc[m], false, false);
      // scheduling hint: issue the 16 DS reads first, then the 8 WMMAs,
      // so waits are graded and the matrix pipe runs back-to-back.
      __builtin_amdgcn_sched_group_barrier(0x100, 16, 0);  // DS reads
      __builtin_amdgcn_sched_group_barrier(0x008,  8, 0);  // WMMA
    }

    if (haveNext) {                      // stage next tile into other buffer
      __bf16* sAn = sm.a[(s + 1) & 1] + sAoff;
#pragma unroll
      for (int q = 0; q < 4; ++q) *(v8bf*)(sAn + q * 8) = areg[q];
      bcur[0] = bnext[0];
      bcur[1] = bnext[1];
    }
  }

  // ---- epilogue: gates fused on-chip, two halves of the M tile ------------
  float* hout = out;
  float* cout = out + (size_t)BATCH * 2048;
  const int zro = (lane >> 4) * 8;       // C/D layout: lanes 16-31 -> rows M+8
  const int zc  = lane & 15;

  for (int half = 0; half < 2; ++half) {
    __syncthreads();
#pragma unroll
    for (int mm = 0; mm < 4; ++mm) {
      float* zt = sm.z + (w * 4 + mm) * 256;
      v8f a = acc[half * 4 + mm];
#pragma unroll
      for (int v = 0; v < 8; ++v)
        zt[(v + zro) * 16 + zc] = a[v];
    }
    __syncthreads();

    const int pair = t >> 1;
    const int row  = pair & 63;          // row within this half
    const int ri2  = pair >> 6;
    const int jb   = (t & 1) * 8;
    const int mm   = row >> 4, rrow = row & 15;
    const float* bias = ri2 ? ibias : rbias;
    const size_t obase = (size_t)(m0 + half * 64 + row) * 2048 + ri2 * 1024 + j0 + jb;

#pragma unroll
    for (int jj = 0; jj < 8; ++jj) {
      int j = jb + jj;
      float z[4];
#pragma unroll
      for (int g = 0; g < 4; ++g)
        z[g] = sm.z[(((ri2 * 4 + g) * 4 + mm) * 16 + rrow) * 16 + j]
             + bias[g * 1024 + j0 + j];
      float iv = fminf(fmaxf(0.2f * z[0] + 0.5f, 0.0f), 1.0f);
      float fv = fminf(fmaxf(0.2f * z[1] + 0.5f, 0.0f), 1.0f);
      float ov = fminf(fmaxf(0.2f * z[3] + 0.5f, 0.0f), 1.0f);
      float cn = fv * c_tm1[obase + jj] + iv * tanhf(z[2]);
      hout[obase + jj] = ov * tanhf(cn);
      cout[obase + jj] = cn;
    }
  }
}

// ---------------------------------------------------------------------------
extern "C" void kernel_launch(void* const* d_in, const int* in_sizes, int n_in,
                              void* d_out, int out_size, void* d_ws, size_t ws_size,
                              hipStream_t stream) {
  (void)in_sizes; (void)n_in; (void)out_size; (void)ws_size;
  const float* inputs = (const float*)d_in[0];
  const float* h_tm1  = (const float*)d_in[1];
  const float* c_tm1  = (const float*)d_in[2];
  const float* rk     = (const float*)d_in[3];
  const float* ik     = (const float*)d_in[4];
  const float* rrk    = (const float*)d_in[5];
  const float* irk    = (const float*)d_in[6];
  const float* rb     = (const float*)d_in[7];
  const float* ib     = (const float*)d_in[8];

  __bf16* Xb = (__bf16*)d_ws;
  __bf16* Wt = (__bf16*)((char*)d_ws + (size_t)BATCH * DK * sizeof(__bf16));

  pack_x<<<(BATCH * (DK / 4)) / 256, 256, 0, stream>>>(inputs, h_tm1, Xb);
  dim3 gw(DK / 32, 8192 / 32);
  pack_wt<<<gw, 256, 0, stream>>>(rk, ik, rrk, irk, Wt);
  dim3 gg(BATCH / BM, 1024 / 16);
  clstm_gemm<<<gg, 256, 0, stream>>>(Xb, Wt, c_tm1, rb, ib, (float*)d_out);
}
